// DecoderLSTM_70635032150029
// MI455X (gfx1250) — matrix operand order repaired and checked by reading, not compiled, not run on hardware
//
#include <hip/hip_runtime.h>

// ---------------------------------------------------------------------------
// DecoderLSTM for MI455X (gfx1250): bf16 WMMA GEMMs, fp32 accumulate.
// Weights pre-swizzled into per-wave 16x32 B-fragment tiles (L2-resident);
// B tiles staged into LDS with GLOBAL_LOAD_ASYNC_TO_LDS_B128 (ASYNCcnt),
// double-buffered, consumed via ds_load_b128 into v_wmma_f32_16x16x32_bf16.
// ---------------------------------------------------------------------------

typedef __attribute__((ext_vector_type(16))) __bf16 v16bf;
typedef __attribute__((ext_vector_type(8)))  float  v8f;

static constexpr int Bm = 512;   // batch
static constexpr int Dm = 64;    // io dim
static constexpr int Em = 512;   // embedding
static constexpr int Hm = 1024;  // hidden
static constexpr int Tm = 96;    // forecast window (reference constant)
static constexpr float LN_EPS = 1e-5f;

union Frag {
    uint4  u[2];
    v16bf  v;
};

__device__ __forceinline__ void async_copy_b128(const void* gsrc, void* ldst)
{
    // LDS byte offset = low 32 bits of the shared-aperture generic pointer.
    const unsigned           dst = (unsigned)(uintptr_t)ldst;
    const unsigned long long src = (unsigned long long)(uintptr_t)gsrc;
    asm volatile("global_load_async_to_lds_b128 %0, %1, off"
                 :: "v"(dst), "v"(src) : "memory");
}

__device__ __forceinline__ void wait_async0()
{
#if __has_builtin(__builtin_amdgcn_s_wait_asynccnt)
    __builtin_amdgcn_s_wait_asynccnt(0);
#else
    asm volatile("s_wait_asynccnt 0x0" ::: "memory");
#endif
}

// ---------------------------------------------------------------------------
// One K-segment of the GEMM: C += A(row-major bf16) x B(packed bf16 tiles).
// Block = 256 threads = 8 waves; per-wave tile = 16(M) x 64(N).
// Stage = 2 K-tiles (K=64) of the block's 4 N-tiles = 8 KB, double-buffered
// in LDS (16 KB). All 8 waves share the staged B; per-lane fragment reads are
// two contiguous ds_load_b128 (packed layout == LDS layout, conflict-free).
// ---------------------------------------------------------------------------
__device__ __forceinline__ void mm_segment(
    const __bf16* __restrict__ A, int lda,
    const __bf16* __restrict__ Bq, int nkbB, int nb0,
    uint4* sB, int tid, int lane, int aHi, int mRow,
    v8f acc[4])
{
    const int nseg = nkbB >> 1;      // two 16x32 K-tiles per stage

    // async stage loader: 8 KB per stage, 2 x async b128 per thread
    auto stage = [&](int seg, int buf) {
        #pragma unroll
        for (int c0 = 0; c0 < 2; ++c0) {
            const int c   = tid + (c0 << 8);   // 0..511 16-byte chunks
            const int tis = c >> 6;            // 0..7 = (j<<1)|kbo
            const int wi  = c & 63;            // chunk within 1 KB tile
            const int j   = tis >> 1;
            const int kbo = tis & 1;
            const __bf16* src =
                Bq + ((size_t)(nb0 + j) * nkbB + (seg * 2 + kbo)) * 512 + wi * 8;
            async_copy_b128(src, sB + (buf << 9) + (tis << 6) + wi);
        }
    };

    __syncthreads();                 // protect LDS reuse across segments
    stage(0, 0);

    const __bf16* ap = A + (size_t)mRow * lda + aHi * 16;

    for (int seg = 0; seg < nseg; ++seg) {
        wait_async0();               // own async stores to LDS complete
        __syncthreads();             // stage `seg` visible, buf (seg+1)&1 free
        if (seg + 1 < nseg) stage(seg + 1, (seg + 1) & 1);
        const int buf = seg & 1;

        #pragma unroll
        for (int kbo = 0; kbo < 2; ++kbo) {
            // ---- A fragment: 16x32 bf16 (two 16-B global loads) ----
            Frag a;
            a.u[0] = *(const uint4*)(ap);
            a.u[1] = *(const uint4*)(ap + 8);
            ap += 32;

            // ---- preload all 4 B fragments, then 4 back-to-back WMMAs ----
            Frag b[4];
            #pragma unroll
            for (int j = 0; j < 4; ++j) {
                const uint4* bp =
                    sB + (buf << 9) + ((((j << 1) | kbo)) << 6) + (lane << 1);
                b[j].u[0] = bp[0];   // ds_load_b128
                b[j].u[1] = bp[1];   // ds_load_b128
            }
            #pragma unroll
            for (int j = 0; j < 4; ++j)
                acc[j] = __builtin_amdgcn_wmma_f32_16x16x32_bf16(
                    false, a.v, false, b[j].v, (short)0, acc[j], false, false);
        }
    }
}

// C[M x N] = [A1 | A2] x [B1 ; B2] + bias ; optional fp32 / bf16 stores.
__global__ __launch_bounds__(256)
void gemm_bf16_wmma(const __bf16* __restrict__ A1, int K1,
                    const __bf16* __restrict__ A2, int K2,
                    const __bf16* __restrict__ B1,
                    const __bf16* __restrict__ B2,
                    const float*  __restrict__ bias,
                    float*        __restrict__ Cf, int ldc, int coff,
                    __bf16*       __restrict__ Cb, int ldcb)
{
    __shared__ uint4 sB[1024];       // 16 KB double buffer

    const int tid   = threadIdx.x;
    const int lane  = tid & 31;
    const int wave  = tid >> 5;
    const int mBase = blockIdx.y * 128 + wave * 16;
    const int nBase = blockIdx.x * 64;
    const int nb0   = nBase >> 4;
    const int aHi   = lane >> 4;
    const int mRow  = mBase + (lane & 15);

    v8f acc[4] = {};

    mm_segment(A1, K1, B1, K1 >> 5, nb0, sB, tid, lane, aHi, mRow, acc);
    if (K2 > 0)                      // uniform across the block
        mm_segment(A2, K2, B2, K2 >> 5, nb0, sB, tid, lane, aHi, mRow, acc);

    // ---- epilogue: bias add + stores ----
    // C layout: VGPR r, lanes 0-15 -> M=r, N=lane; lanes 16-31 -> M=8+r.
    const int cRow0 = mBase + aHi * 8;
    const int cCol  = lane & 15;
    if (Cf) {
        #pragma unroll
        for (int j = 0; j < 4; ++j) {
            const int   col = nBase + j * 16 + cCol;
            const float bv  = bias[col];
            #pragma unroll
            for (int r = 0; r < 8; ++r)
                Cf[(size_t)(cRow0 + r) * ldc + coff + col] = acc[j][r] + bv;
        }
    }
    if (Cb) {
        #pragma unroll
        for (int j = 0; j < 4; ++j) {
            const int   col = nBase + j * 16 + cCol;
            const float bv  = bias[col];
            #pragma unroll
            for (int r = 0; r < 8; ++r)
                Cb[(size_t)(cRow0 + r) * ldcb + col] = (__bf16)(acc[j][r] + bv);
        }
    }
}

// ---------------------------------------------------------------------------
// Pack fp32 weight [K x N row-major] into bf16 WMMA B-fragment tiles.
// P[tile*512 + lane*16 + e] = W[(kb*32 + (lane>>4)*16 + e)*N + nb*16 + (lane&15)]
// with tile = nb*(K/32) + kb  (K-contiguous stream per column block).
// ---------------------------------------------------------------------------
__global__ void pack_w_bf16(const float* __restrict__ W,
                            __bf16* __restrict__ P, int K, int N)
{
    const size_t idx   = (size_t)blockIdx.x * blockDim.x + threadIdx.x;
    const size_t total = (size_t)K * N;
    if (idx >= total) return;
    const int    e    = idx & 15;
    const int    lane = (idx >> 4) & 31;
    const size_t tile = idx >> 9;
    const int    nkb  = K >> 5;
    const int    nb   = (int)(tile / nkb);
    const int    kb   = (int)(tile % nkb);
    const int    k    = (kb << 5) + ((lane >> 4) << 4) + e;
    const int    n    = (nb << 4) + (lane & 15);
    P[idx] = (__bf16)W[(size_t)k * N + n];
}

__global__ void cvt_f32_to_bf16(const float* __restrict__ s,
                                __bf16* __restrict__ d, int n)
{
    const int i = blockIdx.x * blockDim.x + threadIdx.x;
    if (i < n) d[i] = (__bf16)s[i];
}

__global__ void copy_f32(const float* __restrict__ s,
                         float* __restrict__ d, int n)
{
    const int i = blockIdx.x * blockDim.x + threadIdx.x;
    if (i < n) d[i] = s[i];
}

__global__ void add_bias_pair(const float* __restrict__ a,
                              const float* __restrict__ b,
                              float* __restrict__ o, int n)
{
    const int i = blockIdx.x * blockDim.x + threadIdx.x;
    if (i < n) o[i] = a[i] + b[i];
}

// gates [B x 4H] (i,f,g,o chunks) + c -> c (fp32, in ws) and h (bf16)
__global__ void lstm_pointwise(const float* __restrict__ gates,
                               float* __restrict__ c,
                               __bf16* __restrict__ hb)
{
    const int i = blockIdx.x * blockDim.x + threadIdx.x;
    if (i >= Bm * Hm) return;
    const int b = i / Hm, j = i - b * Hm;
    const float* g = gates + (size_t)b * (4 * Hm);
    const float gi = g[j];
    const float gf = g[Hm + j];
    const float gg = g[2 * Hm + j];
    const float go = g[3 * Hm + j];
    const float si = 1.f / (1.f + __expf(-gi));
    const float sf = 1.f / (1.f + __expf(-gf));
    const float so = 1.f / (1.f + __expf(-go));
    const float c2 = sf * c[i] + si * tanhf(gg);
    c[i]  = c2;
    hb[i] = (__bf16)(so * tanhf(c2));
}

// z [B x H] -> LayerNorm -> ReLU -> bf16 out ; one block per row
__global__ __launch_bounds__(256)
void ln_relu(const float* __restrict__ z,
             const float* __restrict__ g,
             const float* __restrict__ bt,
             __bf16* __restrict__ out)
{
    __shared__ float s1[256], s2[256];
    const int row = blockIdx.x;
    const float* zr = z + (size_t)row * Hm;
    float s = 0.f, ss = 0.f;
    for (int j = threadIdx.x; j < Hm; j += 256) {
        const float v = zr[j];
        s += v; ss += v * v;
    }
    s1[threadIdx.x] = s; s2[threadIdx.x] = ss;
    __syncthreads();
    for (int k = 128; k > 0; k >>= 1) {
        if (threadIdx.x < k) {
            s1[threadIdx.x] += s1[threadIdx.x + k];
            s2[threadIdx.x] += s2[threadIdx.x + k];
        }
        __syncthreads();
    }
    const float mu  = s1[0] * (1.f / Hm);
    const float var = s2[0] * (1.f / Hm) - mu * mu;
    const float inv = rsqrtf(var + LN_EPS);
    for (int j = threadIdx.x; j < Hm; j += 256) {
        const float v = (zr[j] - mu) * inv * g[j] + bt[j];
        out[(size_t)row * Hm + j] = (__bf16)fmaxf(v, 0.f);
    }
}

// ---------------------------------------------------------------------------
extern "C" void kernel_launch(void* const* d_in, const int* in_sizes, int n_in,
                              void* d_out, int out_size, void* d_ws, size_t ws_size,
                              hipStream_t stream)
{
    const float* x_0   = (const float*)d_in[0];
    const float* h_n   = (const float*)d_in[1];
    const float* c_n   = (const float*)d_in[2];
    /* d_in[3] forecast_window: device scalar, use compile-time Tm */
    const float* emb_W = (const float*)d_in[4];
    const float* emb_b = (const float*)d_in[5];
    const float* W_ih0 = (const float*)d_in[6];
    const float* W_hh0 = (const float*)d_in[7];
    const float* b_ih0 = (const float*)d_in[8];
    const float* b_hh0 = (const float*)d_in[9];
    const float* W_ih1 = (const float*)d_in[10];
    const float* W_hh1 = (const float*)d_in[11];
    const float* b_ih1 = (const float*)d_in[12];
    const float* b_hh1 = (const float*)d_in[13];
    const float* fc1_W = (const float*)d_in[14];
    const float* fc1_b = (const float*)d_in[15];
    const float* ln_g  = (const float*)d_in[16];
    const float* ln_b  = (const float*)d_in[17];
    const float* fc2_W = (const float*)d_in[18];
    const float* fc2_b = (const float*)d_in[19];
    float* out = (float*)d_out;

    // ---- workspace bump allocator (256-B aligned) ----
    char* wp = (char*)d_ws;
    auto alloc = [&](size_t bytes) -> char* {
        char* p = wp;
        wp += (bytes + 255) & ~(size_t)255;
        return p;
    };
    __bf16* P_emb = (__bf16*)alloc((size_t)Dm * Em * 2);
    __bf16* P_ih0 = (__bf16*)alloc((size_t)Em * 4 * Hm * 2);
    __bf16* P_hh0 = (__bf16*)alloc((size_t)Hm * 4 * Hm * 2);
    __bf16* P_ih1 = (__bf16*)alloc((size_t)Hm * 4 * Hm * 2);
    __bf16* P_hh1 = (__bf16*)alloc((size_t)Hm * 4 * Hm * 2);
    __bf16* P_fc1 = (__bf16*)alloc((size_t)Hm * Hm * 2);
    __bf16* P_fc2 = (__bf16*)alloc((size_t)Hm * Dm * 2);
    float*  b0    = (float*)alloc(4 * Hm * 4);
    float*  b1    = (float*)alloc(4 * Hm * 4);
    __bf16* x_bf  = (__bf16*)alloc((size_t)Bm * Dm * 2);
    __bf16* xe_bf = (__bf16*)alloc((size_t)Bm * Em * 2);
    __bf16* h0_bf = (__bf16*)alloc((size_t)Bm * Hm * 2);
    __bf16* h1_bf = (__bf16*)alloc((size_t)Bm * Hm * 2);
    float*  c0    = (float*)alloc((size_t)Bm * Hm * 4);
    float*  c1    = (float*)alloc((size_t)Bm * Hm * 4);
    float*  gates = (float*)alloc((size_t)Bm * 4 * Hm * 4);
    float*  zbuf  = (float*)alloc((size_t)Bm * Hm * 4);
    __bf16* zr_bf = (__bf16*)alloc((size_t)Bm * Hm * 2);
    (void)ws_size; (void)in_sizes; (void)n_in; (void)out_size;

    auto blocks = [](size_t n) { return (unsigned)((n + 255) / 256); };

    // ---- one-time (per launch) weight packing + state init ----
    pack_w_bf16<<<blocks((size_t)Dm * Em), 256, 0, stream>>>(emb_W, P_emb, Dm, Em);
    pack_w_bf16<<<blocks((size_t)Em * 4 * Hm), 256, 0, stream>>>(W_ih0, P_ih0, Em, 4 * Hm);
    pack_w_bf16<<<blocks((size_t)Hm * 4 * Hm), 256, 0, stream>>>(W_hh0, P_hh0, Hm, 4 * Hm);
    pack_w_bf16<<<blocks((size_t)Hm * 4 * Hm), 256, 0, stream>>>(W_ih1, P_ih1, Hm, 4 * Hm);
    pack_w_bf16<<<blocks((size_t)Hm * 4 * Hm), 256, 0, stream>>>(W_hh1, P_hh1, Hm, 4 * Hm);
    pack_w_bf16<<<blocks((size_t)Hm * Hm), 256, 0, stream>>>(fc1_W, P_fc1, Hm, Hm);
    pack_w_bf16<<<blocks((size_t)Hm * Dm), 256, 0, stream>>>(fc2_W, P_fc2, Hm, Dm);
    add_bias_pair<<<blocks(4 * Hm), 256, 0, stream>>>(b_ih0, b_hh0, b0, 4 * Hm);
    add_bias_pair<<<blocks(4 * Hm), 256, 0, stream>>>(b_ih1, b_hh1, b1, 4 * Hm);
    cvt_f32_to_bf16<<<blocks((size_t)Bm * Dm), 256, 0, stream>>>(x_0, x_bf, Bm * Dm);
    cvt_f32_to_bf16<<<blocks((size_t)Bm * Hm), 256, 0, stream>>>(h_n, h0_bf, Bm * Hm);
    cvt_f32_to_bf16<<<blocks((size_t)Bm * Hm), 256, 0, stream>>>(h_n + (size_t)Bm * Hm, h1_bf, Bm * Hm);
    copy_f32<<<blocks((size_t)Bm * Hm), 256, 0, stream>>>(c_n, c0, Bm * Hm);
    copy_f32<<<blocks((size_t)Bm * Hm), 256, 0, stream>>>(c_n + (size_t)Bm * Hm, c1, Bm * Hm);

    const dim3 blk(256);
    const dim3 g_emb(Em / 64, Bm / 128);      // 512 x 512
    const dim3 g_gts(4 * Hm / 64, Bm / 128);  // 512 x 4096
    const dim3 g_fc1(Hm / 64, Bm / 128);      // 512 x 1024
    const dim3 g_fc2(Dm / 64, Bm / 128);      // 512 x 64

    for (int t = 0; t < Tm; ++t) {
        // xe = x @ emb_W + emb_b  -> bf16
        gemm_bf16_wmma<<<g_emb, blk, 0, stream>>>(
            x_bf, Dm, x_bf, 0, P_emb, P_emb, emb_b,
            nullptr, 0, 0, xe_bf, Em);

        // layer-0 gates = [xe | h0] @ [Wih0 ; Whh0] + b0
        gemm_bf16_wmma<<<g_gts, blk, 0, stream>>>(
            xe_bf, Em, h0_bf, Hm, P_ih0, P_hh0, b0,
            gates, 4 * Hm, 0, nullptr, 0);
        lstm_pointwise<<<blocks((size_t)Bm * Hm), 256, 0, stream>>>(gates, c0, h0_bf);

        // layer-1 gates = [h0 | h1] @ [Wih1 ; Whh1] + b1
        gemm_bf16_wmma<<<g_gts, blk, 0, stream>>>(
            h0_bf, Hm, h1_bf, Hm, P_ih1, P_hh1, b1,
            gates, 4 * Hm, 0, nullptr, 0);
        lstm_pointwise<<<blocks((size_t)Bm * Hm), 256, 0, stream>>>(gates, c1, h1_bf);

        // z = h1 @ fc1_W + fc1_b ; LayerNorm ; ReLU -> bf16
        gemm_bf16_wmma<<<g_fc1, blk, 0, stream>>>(
            h1_bf, Hm, h1_bf, 0, P_fc1, P_fc1, fc1_b,
            zbuf, Hm, 0, nullptr, 0);
        ln_relu<<<Bm, 256, 0, stream>>>(zbuf, ln_g, ln_b, zr_bf);

        // y = relu(z) @ fc2_W + fc2_b -> d_out[:, t, :] (fp32) + x_bf (bf16)
        gemm_bf16_wmma<<<g_fc2, blk, 0, stream>>>(
            zr_bf, Hm, zr_bf, 0, P_fc2, P_fc2, fc2_b,
            out, Tm * Dm, t * Dm, x_bf, Dm);
    }
}
